// Transformer_encoder_block_18202071401004
// MI455X (gfx1250) — matrix operand order repaired and checked by reading, not compiled, not run on hardware
//
#include <hip/hip_runtime.h>
#include <cmath>

// ---------------------------------------------------------------------------
// Transformer encoder block for MI455X (gfx1250, wave32, WMMA).
// B=4, N=2048, D=384, H=6, hd=64, DFF=2048.  fp32 in/out, f16 WMMA inputs,
// fp32 accumulation.  GEMMs: block-cooperative B-tile staging through LDS
// using async global->LDS copies (ASYNCcnt), double-buffered; per-wave 32x64
// output tiles (8 wmmas per k-step).
// ---------------------------------------------------------------------------

typedef _Float16 half_t;
typedef __attribute__((ext_vector_type(8)))  _Float16 v8h;
typedef __attribute__((ext_vector_type(16))) _Float16 v16h;
typedef __attribute__((ext_vector_type(8)))  float    v8f;

// reuse_a hint: set when the NEXT wmma uses the same A operand (VOP3P RA).
#define WMMA_F16(a, b, c, ra) \
  __builtin_amdgcn_wmma_f32_16x16x32_f16(false, (a), false, (b), (short)0, (c), (ra), false)

// ----- async global->LDS copy (16B per lane) -------------------------------
// ISA: GLOBAL_LOAD_ASYNC_TO_LDS_B128  vdst(LDS byte addr), vaddr(64b global),
// off (no saddr).  Tracked by ASYNCcnt.  LDS offset = low 32 bits of the
// generic pointer (flat LDS aperture keeps wave-relative offset in [31:0]).
__device__ __forceinline__ void async_copy_b128(const half_t* src, half_t* dstLds) {
  unsigned int ldsoff = (unsigned int)(unsigned long long)dstLds;
  unsigned long long gaddr = (unsigned long long)src;
  asm volatile("global_load_async_to_lds_b128 %0, %1, off"
               :: "v"(ldsoff), "v"(gaddr) : "memory");
}

template <int N>
__device__ __forceinline__ void wait_async() {
#if __has_builtin(__builtin_amdgcn_s_wait_asynccnt)
  __builtin_amdgcn_s_wait_asynccnt(N);
#else
  asm volatile("s_wait_asynccnt %0" :: "n"(N) : "memory");
#endif
}

// ----- fragment loaders ----------------------------------------------------
// A (16x32 f16) row-major: lanes 0-15 row M=lane, K 0..15 contiguous; lanes
// 16-31 same rows, K 16..31.
__device__ __forceinline__ v16h load_frag_a(const half_t* __restrict__ base,
                                            int row0, int ld, int k0) {
  int lane = threadIdx.x & 31;
  int r = lane & 15, hi = lane >> 4;
  const half_t* p = base + (size_t)(row0 + r) * ld + k0 + hi * 16;
  v8h lo = *(const v8h*)p;
  v8h hv = *(const v8h*)(p + 8);
  return __builtin_shufflevector(lo, hv, 0, 1, 2, 3, 4, 5, 6, 7,
                                 8, 9, 10, 11, 12, 13, 14, 15);
}

// B (32x16 f16) from transposed source BT[N][K]: lanes 0-15 col N=lane,
// K 0..15 contiguous; lanes 16-31 K 16..31.
__device__ __forceinline__ v16h load_frag_b(const half_t* __restrict__ bT,
                                            int col0, int ld, int k0) {
  int lane = threadIdx.x & 31;
  int c = lane & 15, hi = lane >> 4;
  const half_t* p = bT + (size_t)(col0 + c) * ld + k0 + hi * 16;
  v8h lo = *(const v8h*)p;
  v8h hv = *(const v8h*)(p + 8);
  return __builtin_shufflevector(lo, hv, 0, 1, 2, 3, 4, 5, 6, 7,
                                 8, 9, 10, 11, 12, 13, 14, 15);
}

// B fragment from an LDS tile [64 cols][40 halves] (32 k + 8 pad).
__device__ __forceinline__ v16h load_frag_b_lds(const half_t* sbuf, int colt) {
  int lane = threadIdx.x & 31;
  int c = lane & 15, hi = lane >> 4;
  const half_t* p = sbuf + (colt + c) * 40 + hi * 16;
  v8h lo = *(const v8h*)p;
  v8h hv = *(const v8h*)(p + 8);
  return __builtin_shufflevector(lo, hv, 0, 1, 2, 3, 4, 5, 6, 7,
                                 8, 9, 10, 11, 12, 13, 14, 15);
}

// stage one 64x32 B tile into LDS: 256 threads x one b128 async copy each.
__device__ __forceinline__ void stage_b_tile(const half_t* __restrict__ BT, int ldb,
                                             int col0, int k0, half_t* sbuf) {
  int t = threadIdx.x;          // blockDim.x == 256
  int col = t >> 2, seg = t & 3;
  const half_t* src = BT + (size_t)(col0 + col) * ldb + k0 + seg * 8;
  async_copy_b128(src, sbuf + col * 40 + seg * 8);
}

// ----- block-cooperative GEMM core: block = 256 rows x 64 cols -------------
// Each of 8 waves computes a 32x64 tile: acc[2][4].
__device__ __forceinline__ void gemm_block_32x64(
    const half_t* __restrict__ A, int lda,
    const half_t* __restrict__ BT, int ldb, int K,
    int rowW, int col0, half_t* sB /* [2][64*40] */, v8f acc[2][4]) {
  const int KT = K >> 5;
  stage_b_tile(BT, ldb, col0, 0, sB);
  for (int kt = 0; kt < KT; ++kt) {
    int cur = kt & 1;
    if (kt + 1 < KT) {
      stage_b_tile(BT, ldb, col0, (kt + 1) * 32, sB + (1 - cur) * (64 * 40));
      wait_async<1>();   // oldest (current tile) done; next still in flight
    } else {
      wait_async<0>();
    }
    __syncthreads();
    const half_t* buf = sB + cur * (64 * 40);
    v16h bf[4];
#pragma unroll
    for (int t = 0; t < 4; ++t) bf[t] = load_frag_b_lds(buf, t * 16);
#pragma unroll
    for (int g = 0; g < 2; ++g) {
      v16h a = load_frag_a(A, rowW + g * 16, lda, kt * 32);
      acc[g][0] = WMMA_F16(a, bf[0], acc[g][0], true);
      acc[g][1] = WMMA_F16(a, bf[1], acc[g][1], true);
      acc[g][2] = WMMA_F16(a, bf[2], acc[g][2], true);
      acc[g][3] = WMMA_F16(a, bf[3], acc[g][3], false);
    }
    __syncthreads();
  }
}

// ----- LayerNorm: one wave per row of 384, f32 in -> f16 out ---------------
__global__ void ln_to_f16_kernel(const float* __restrict__ x,
                                 const float* __restrict__ g,
                                 const float* __restrict__ b,
                                 half_t* __restrict__ y, int rows) {
  int wave = (int)((blockIdx.x * blockDim.x + threadIdx.x) >> 5);
  int lane = threadIdx.x & 31;
  if (wave >= rows) return;
  const float* xr = x + (size_t)wave * 384;
  float v[12];
  float s = 0.f;
#pragma unroll
  for (int i = 0; i < 12; ++i) { v[i] = xr[lane + i * 32]; s += v[i]; }
#pragma unroll
  for (int off = 16; off; off >>= 1) s += __shfl_xor(s, off, 32);
  float mu = s * (1.0f / 384.0f);
  float q = 0.f;
#pragma unroll
  for (int i = 0; i < 12; ++i) { float d = v[i] - mu; q += d * d; }
#pragma unroll
  for (int off = 16; off; off >>= 1) q += __shfl_xor(q, off, 32);
  float rs = rsqrtf(q * (1.0f / 384.0f) + 1e-5f);
  half_t* yr = y + (size_t)wave * 384;
#pragma unroll
  for (int i = 0; i < 12; ++i) {
    int c = lane + i * 32;
    yr[c] = (half_t)((v[i] - mu) * rs * g[c] + b[c]);
  }
}

// ----- weight transpose + f16 convert: in[K][N] f32 -> out[N][K] f16 -------
__global__ void transpose_to_f16_kernel(const float* __restrict__ in,
                                        half_t* __restrict__ out,
                                        int K, int N) {
  size_t tid = (size_t)blockIdx.x * blockDim.x + threadIdx.x;
  if (tid >= (size_t)K * N) return;
  int n = (int)(tid / K), k = (int)(tid % K);
  out[tid] = (half_t)in[(size_t)k * N + n];
}

// ----- QKV GEMM: [8192,384]x[384,1152] -> scatter q,k (row-major) & vT -----
__global__ void __launch_bounds__(256)
qkv_gemm_kernel(const half_t* __restrict__ A, const half_t* __restrict__ BT,
                const float* __restrict__ bias,
                half_t* __restrict__ q, half_t* __restrict__ k,
                half_t* __restrict__ vT) {
  const int NT = 1152 / 64;  // 18
  __shared__ __attribute__((aligned(16))) half_t sB[2][64 * 40];
  int mb = blockIdx.x / NT, nb = blockIdx.x % NT;
  int wave = threadIdx.x >> 5, lane = threadIdx.x & 31;
  int rowW = mb * 256 + wave * 32, col0 = nb * 64;
  v8f acc[2][4] = {};
  gemm_block_32x64(A, 384, BT, 384, 384, rowW, col0, &sB[0][0], acc);
  int r16 = lane & 15, hi = lane >> 4;
#pragma unroll
  for (int g = 0; g < 2; ++g) {
#pragma unroll
    for (int t = 0; t < 4; ++t) {
      int col = col0 + t * 16 + r16;  // 16-col groups never straddle 64/384 bounds
      int which = col / 384;
      int d = col % 384;
      int head = d >> 6, n = d & 63;
      float bv = bias[col];
#pragma unroll
      for (int r = 0; r < 8; ++r) {
        int row = rowW + g * 16 + r + hi * 8;
        int bb = row >> 11, nn = row & 2047;
        size_t bh = (size_t)(bb * 6 + head);
        float val = acc[g][t][r] + bv;
        if (which == 0)      q[(bh * 2048 + nn) * 64 + n] = (half_t)val;
        else if (which == 1) k[(bh * 2048 + nn) * 64 + n] = (half_t)val;
        else                 vT[(bh * 64 + n) * 2048 + nn] = (half_t)val;
      }
    }
  }
}

// ----- Flash attention: wave owns 16 query rows, streams 32 keys/step ------
__global__ void attn_kernel(const half_t* __restrict__ q,
                            const half_t* __restrict__ k,
                            const half_t* __restrict__ vT,
                            const float* __restrict__ x,
                            float* __restrict__ x2) {
  const float scale = 0.125f;  // 1/sqrt(64)
  int wavein = threadIdx.x >> 5;
  int lane = threadIdx.x & 31;
  int gw = blockIdx.x * (blockDim.x >> 5) + wavein;
  if (gw >= 24 * 128) return;
  int bh = gw >> 7, rb = gw & 127;
  int row0 = rb * 16;
  const half_t* qb = q + (size_t)bh * 2048 * 64;
  const half_t* kb = k + (size_t)bh * 2048 * 64;
  const half_t* vb = vT + (size_t)bh * 64 * 2048;

  __shared__ __attribute__((aligned(16))) half_t ptile[4][16 * 32];
  half_t* P = ptile[wavein];

  v16h qa0 = load_frag_a(qb, row0, 64, 0);
  v16h qa1 = load_frag_a(qb, row0, 64, 32);

  v8f o[4] = {};
  float m[8], l[8];
#pragma unroll
  for (int r = 0; r < 8; ++r) { m[r] = -3.0e38f; l[r] = 0.f; }
  int r16 = lane & 15, hi = lane >> 4;

  for (int j0 = 0; j0 < 2048; j0 += 32) {
    v8f s0 = {}, s1 = {};
    {
      v16h b0 = load_frag_b(kb, j0, 64, 0);
      v16h b1 = load_frag_b(kb, j0, 64, 32);
      s0 = WMMA_F16(qa0, b0, s0, false);
      s0 = WMMA_F16(qa1, b1, s0, false);
      v16h b2 = load_frag_b(kb, j0 + 16, 64, 0);
      v16h b3 = load_frag_b(kb, j0 + 16, 64, 32);
      s1 = WMMA_F16(qa0, b2, s1, false);
      s1 = WMMA_F16(qa1, b3, s1, false);
    }
    float alpha[8];
#pragma unroll
    for (int r = 0; r < 8; ++r) {
      float a = fmaxf(s0[r], s1[r]) * scale;
#pragma unroll
      for (int off = 1; off < 16; off <<= 1) a = fmaxf(a, __shfl_xor(a, off, 32));
      float mn = fmaxf(m[r], a);
      alpha[r] = __expf(m[r] - mn);
      m[r] = mn;
      float p0 = __expf(s0[r] * scale - mn);
      float p1 = __expf(s1[r] * scale - mn);
      P[(r + hi * 8) * 32 + r16]      = (half_t)p0;
      P[(r + hi * 8) * 32 + 16 + r16] = (half_t)p1;
      float cs = p0 + p1;
#pragma unroll
      for (int off = 1; off < 16; off <<= 1) cs += __shfl_xor(cs, off, 32);
      l[r] = l[r] * alpha[r] + cs;
    }
#pragma unroll
    for (int t = 0; t < 4; ++t)
#pragma unroll
      for (int r = 0; r < 8; ++r) o[t][r] *= alpha[r];

    // wave-private LDS tile: ensure stores landed before transposed re-read
    asm volatile("s_wait_dscnt 0" ::: "memory");
    const half_t* pp = P + r16 * 32 + hi * 16;
    v8h plo = *(const v8h*)pp;
    v8h phi = *(const v8h*)(pp + 8);
    v16h pa = __builtin_shufflevector(plo, phi, 0, 1, 2, 3, 4, 5, 6, 7,
                                      8, 9, 10, 11, 12, 13, 14, 15);
    {
      v16h v0 = load_frag_b(vb, 0, 2048, j0);
      v16h v1 = load_frag_b(vb, 16, 2048, j0);
      v16h v2 = load_frag_b(vb, 32, 2048, j0);
      v16h v3 = load_frag_b(vb, 48, 2048, j0);
      o[0] = WMMA_F16(pa, v0, o[0], true);
      o[1] = WMMA_F16(pa, v1, o[1], true);
      o[2] = WMMA_F16(pa, v2, o[2], true);
      o[3] = WMMA_F16(pa, v3, o[3], false);
    }
  }

  float invl[8];
#pragma unroll
  for (int r = 0; r < 8; ++r) invl[r] = 1.0f / l[r];
  int b = bh / 6, h = bh % 6;
#pragma unroll
  for (int t = 0; t < 4; ++t) {
    int n = t * 16 + r16;
#pragma unroll
    for (int r = 0; r < 8; ++r) {
      int row = row0 + r + hi * 8;
      size_t idx = ((size_t)(b * 2048) + row) * 384 + h * 64 + n;
      x2[idx] = x[idx] + o[t][r] * invl[r];
    }
  }
}

// ----- MLP1: [8192,384]x[384,2048], exact GELU, f16 out --------------------
__global__ void __launch_bounds__(256)
mlp1_gemm_kernel(const half_t* __restrict__ A, const half_t* __restrict__ BT,
                 const float* __restrict__ bias, half_t* __restrict__ H) {
  const int NT = 2048 / 64;  // 32
  __shared__ __attribute__((aligned(16))) half_t sB[2][64 * 40];
  int mb = blockIdx.x / NT, nb = blockIdx.x % NT;
  int wave = threadIdx.x >> 5, lane = threadIdx.x & 31;
  int rowW = mb * 256 + wave * 32, col0 = nb * 64;
  v8f acc[2][4] = {};
  gemm_block_32x64(A, 384, BT, 384, 384, rowW, col0, &sB[0][0], acc);
  int r16 = lane & 15, hi = lane >> 4;
#pragma unroll
  for (int g = 0; g < 2; ++g) {
#pragma unroll
    for (int t = 0; t < 4; ++t) {
      int col = col0 + t * 16 + r16;
      float bv = bias[col];
#pragma unroll
      for (int r = 0; r < 8; ++r) {
        int row = rowW + g * 16 + r + hi * 8;
        float v = acc[g][t][r] + bv;
        float gv = 0.5f * v * (1.0f + erff(v * 0.70710678118f));  // exact GELU
        H[(size_t)row * 2048 + col] = (half_t)gv;
      }
    }
  }
}

// ----- MLP2: [8192,2048]x[2048,384] + residual, fp32 out -------------------
__global__ void __launch_bounds__(256)
mlp2_gemm_kernel(const half_t* __restrict__ A, const half_t* __restrict__ BT,
                 const float* __restrict__ bias, const float* __restrict__ x2,
                 float* __restrict__ out) {
  const int NT = 384 / 64;  // 6
  __shared__ __attribute__((aligned(16))) half_t sB[2][64 * 40];
  int mb = blockIdx.x / NT, nb = blockIdx.x % NT;
  int wave = threadIdx.x >> 5, lane = threadIdx.x & 31;
  int rowW = mb * 256 + wave * 32, col0 = nb * 64;
  v8f acc[2][4] = {};
  gemm_block_32x64(A, 2048, BT, 2048, 2048, rowW, col0, &sB[0][0], acc);
  int r16 = lane & 15, hi = lane >> 4;
#pragma unroll
  for (int g = 0; g < 2; ++g) {
#pragma unroll
    for (int t = 0; t < 4; ++t) {
      int col = col0 + t * 16 + r16;
      float bv = bias[col];
#pragma unroll
      for (int r = 0; r < 8; ++r) {
        int row = rowW + g * 16 + r + hi * 8;
        size_t idx = (size_t)row * 384 + col;
        out[idx] = x2[idx] + acc[g][t][r] + bv;
      }
    }
  }
}

// ---------------------------------------------------------------------------
extern "C" void kernel_launch(void* const* d_in, const int* in_sizes, int n_in,
                              void* d_out, int out_size, void* d_ws, size_t ws_size,
                              hipStream_t stream) {
  (void)in_sizes; (void)n_in; (void)out_size; (void)ws_size;
  const float* x      = (const float*)d_in[0];
  const float* norm_g = (const float*)d_in[1];
  const float* norm_b = (const float*)d_in[2];
  const float* w_qkv  = (const float*)d_in[3];
  const float* b_qkv  = (const float*)d_in[4];
  const float* w1     = (const float*)d_in[5];
  const float* b1     = (const float*)d_in[6];
  const float* w2     = (const float*)d_in[7];
  const float* b2     = (const float*)d_in[8];
  float* out = (float*)d_out;

  // workspace layout (~57 MB; q/k/vT aliased over h16 since attention
  // completes before MLP1 starts)
  char* ws = (char*)d_ws;
  size_t off = 0;
  auto alloc = [&](size_t bytes) -> char* {
    char* p = ws + off;
    off += (bytes + 255) & ~(size_t)255;
    return p;
  };
  half_t* wqkvT = (half_t*)alloc((size_t)1152 * 384 * 2);
  half_t* w1T   = (half_t*)alloc((size_t)2048 * 384 * 2);
  half_t* w2T   = (half_t*)alloc((size_t)384 * 2048 * 2);
  half_t* y16   = (half_t*)alloc((size_t)8192 * 384 * 2);   // reused for ln2 out
  float*  x2    = (float*) alloc((size_t)8192 * 384 * 4);
  char*   big   = alloc((size_t)8192 * 2048 * 2);           // max(qkv, h16)
  half_t* q16   = (half_t*)big;
  half_t* k16   = (half_t*)(big + (size_t)24 * 2048 * 64 * 2);
  half_t* vT16  = (half_t*)(big + (size_t)48 * 2048 * 64 * 2);
  half_t* h16   = (half_t*)big;

  // weight transposes (f32 -> f16, [K,N] -> [N,K])
  transpose_to_f16_kernel<<<(1152 * 384 + 255) / 256, 256, 0, stream>>>(w_qkv, wqkvT, 384, 1152);
  transpose_to_f16_kernel<<<(2048 * 384 + 255) / 256, 256, 0, stream>>>(w1, w1T, 384, 2048);
  transpose_to_f16_kernel<<<(384 * 2048 + 255) / 256, 256, 0, stream>>>(w2, w2T, 2048, 384);

  // 1) ln1(x) -> y16
  ln_to_f16_kernel<<<1024, 256, 0, stream>>>(x, norm_g, norm_b, y16, 8192);
  // 2) qkv = y16 @ wqkv + b -> q,k (row-major), v (transposed)
  qkv_gemm_kernel<<<32 * 18, 256, 0, stream>>>(y16, wqkvT, b_qkv, q16, k16, vT16);
  // 3) flash attention, x2 = x + att
  attn_kernel<<<(24 * 128) / 4, 128, 0, stream>>>(q16, k16, vT16, x, x2);
  // 4) ln2(x2) -> y16 (reuse)
  ln_to_f16_kernel<<<1024, 256, 0, stream>>>(x2, norm_g, norm_b, y16, 8192);
  // 5) h = gelu(y16 @ w1 + b1)
  mlp1_gemm_kernel<<<32 * 32, 256, 0, stream>>>(y16, w1T, b1, h16);
  // 6) out = x2 + h @ w2 + b2
  mlp2_gemm_kernel<<<32 * 6, 256, 0, stream>>>(h16, w2T, b2, x2, out);
}
